// CLloss_43714177138893
// MI455X (gfx1250) — compile-verified, matrix operand order checked by reading, not run on hardware
//
#include <hip/hip_runtime.h>

#define MM 8192
#define CC 4096
#define DD 768
#define BM 64            // rows per workgroup
#define BN 32            // cols per tile
#define KC 32            // K per WMMA
#define NKC (DD / KC)    // 24
#define NCT (CC / BN)    // 128
#define BROW_PAD 776     // 768 + 8 bf16 elements (bank-spread padding, keeps 16B align)
#define BBUF_BYTES (BN * BROW_PAD * 2)
#define EPSV 1e-8f

typedef __attribute__((ext_vector_type(16))) __bf16 v16bf;
typedef __attribute__((ext_vector_type(8)))  __bf16 v8bf;
typedef __attribute__((ext_vector_type(8)))  float  v8f;

// ---------------- Prep: fused alpha-scale + exp_map0 + time component ----------------
__global__ void __launch_bounds__(256)
prep_kernel(const float* __restrict__ text,
            const float* __restrict__ label,
            const float* __restrict__ curv_log,
            const float* __restrict__ text_alpha_log,
            const float* __restrict__ label_alpha_log,
            __bf16* __restrict__ textHyp,
            __bf16* __restrict__ labelHyp,
            float* __restrict__ xTime,
            float* __restrict__ yTime)
{
    const int bid = blockIdx.x;
    const int tid = threadIdx.x;
    const bool isText = bid < MM;
    const int row = isText ? bid : bid - MM;
    const float* src = (isText ? text : label) + (size_t)row * DD;
    __bf16* dst = (isText ? textHyp : labelHyp) + (size_t)row * DD;
    float* timeOut = (isText ? xTime : yTime) + row;

    const float curv  = expf(curv_log[0]);
    const float alpha = expf((isText ? text_alpha_log : label_alpha_log)[0]);

    float xs[3];
    float ss = 0.f;
#pragma unroll
    for (int j = 0; j < 3; ++j) {
        float v = src[tid + j * 256] * alpha;
        xs[j] = v;
        ss += v * v;
    }
    // wave32 reduction, then cross-wave via LDS
    for (int o = 16; o > 0; o >>= 1) ss += __shfl_down(ss, o, 32);
    __shared__ float red[8];
    __shared__ float totSh;
    const int lane = tid & 31, wid = tid >> 5;
    if (lane == 0) red[wid] = ss;
    __syncthreads();
    if (tid == 0) {
        float t = 0.f;
        for (int i = 0; i < 8; ++i) t += red[i];
        totSh = t;
    }
    __syncthreads();
    const float n2 = totSh;
    const float rc = sqrtf(curv) * sqrtf(n2);
    const float factor = sinhf(rc) / fmaxf(rc, EPSV);
#pragma unroll
    for (int j = 0; j < 3; ++j)
        dst[tid + j * 256] = (__bf16)(factor * xs[j]);
    if (tid == 0)
        timeOut[0] = sqrtf(1.0f / curv + factor * factor * n2);
}

// Issue 12 per-thread async copies (16B each) of one 32x768 bf16 label strip
// into a padded LDS buffer. No VGPR destinations; tracked by ASYNCcnt.
__device__ __forceinline__ void stage_async(const __bf16* __restrict__ gBase,
                                            __bf16* ldsDst, int tid)
{
    const unsigned long long sa = (unsigned long long)(uintptr_t)gBase;
#pragma unroll
    for (int q = 0; q < 12; ++q) {
        int c  = tid + q * 256;          // 0..3071 chunk id
        int r  = c / 96;                 // label row within tile
        int k8 = c % 96;                 // 8-element column chunk
        unsigned gOff   = (unsigned)((r * DD + k8 * 8) * 2);            // bytes
        unsigned ldsOff = (unsigned)(uintptr_t)(ldsDst + r * BROW_PAD + k8 * 8);
        asm volatile("global_load_async_to_lds_b128 %0, %1, %2"
                     :: "v"(ldsOff), "v"(gOff), "s"(sa)
                     : "memory");
    }
}

// ---------------- Main: WMMA GEMM + fused hard-negative mining + loss ----------------
__global__ void __launch_bounds__(256)
cl_loss_kernel(const __bf16* __restrict__ textHyp,
               const __bf16* __restrict__ labelHyp,
               const float* __restrict__ xTime,
               const float* __restrict__ yTime,
               const int* __restrict__ targets,
               const float* __restrict__ curv_log,
               float* __restrict__ out)
{
    // Double-buffered B strips; candidate-merge array reuses buffer 0 after the loop.
    __shared__ __align__(16) unsigned char smemRaw[2 * BBUF_BYTES];
    __bf16* BsBuf0 = (__bf16*)smemRaw;
    __bf16* BsBuf1 = (__bf16*)(smemRaw + BBUF_BYTES);
    float*  cand   = (float*)smemRaw;          // BM*64 floats = 16KB, used post-loop
    __shared__ float posArr[BM];

    const int tid  = threadIdx.x;
    const int lane = tid & 31;
    const int wave = tid >> 5;
    const int wr = wave >> 1;                 // 0..3 (row wave)
    const int wc = wave & 1;                  // 0..1 (col wave)
    const int half = (lane >= 16) ? 1 : 0;
    const int l16 = lane & 15;
    const int rowBase = blockIdx.x * BM;

    const float curv = expf(curv_log[0]);

    if (tid < BM) posArr[tid] = 0.0f;

    // Per-lane row metadata for the 8 C/D rows this lane owns (M = v + 8*half)
    float xt[8]; int tg[8];
#pragma unroll
    for (int v = 0; v < 8; ++v) {
        int r = rowBase + wr * 16 + v + half * 8;
        xt[v] = xTime[r];
        tg[v] = targets[r];
    }

    float max1[8], max2[8];
#pragma unroll
    for (int v = 0; v < 8; ++v) { max1[v] = -3.0e38f; max2[v] = -3.0e38f; }

    // Load this wave's ENTIRE 16x768 bf16 A-strip into registers once
    // (24 x v16bf = 192 VGPRs/lane, lives in the high VGPR banks via
    // s_set_vgpr_msb). Held as SSA values, it is immune to the asm
    // "memory" clobbers below — no reloads inside the column loop.
    // 16-bit A layout = two contiguous 8-elem chunks per lane at
    // K + {0,16} (+8 for the upper half-wave).
    const __bf16* aBase = textHyp + (size_t)(rowBase + wr * 16 + l16) * DD + half * 8;
    v16bf aFrag[NKC];
#pragma unroll
    for (int kc = 0; kc < NKC; ++kc) {
        v8bf a0 = *(const v8bf*)(aBase + kc * KC);
        v8bf a1 = *(const v8bf*)(aBase + kc * KC + 16);
        aFrag[kc] = __builtin_shufflevector(a0, a1,
                        0,1,2,3,4,5,6,7,8,9,10,11,12,13,14,15);
    }

    // Prime the async pipeline with tile 0.
    stage_async(labelHyp, BsBuf0, tid);

    for (int ct = 0; ct < NCT; ++ct) {
        // Kick off the DMA for the next tile, then wait only for the previous
        // group (async loads complete in order: 12 newer may remain in flight).
        if (ct + 1 < NCT) {
            stage_async(labelHyp + (size_t)(ct + 1) * BN * DD,
                        ((ct + 1) & 1) ? BsBuf1 : BsBuf0, tid);
            if (ct + 2 < NCT)
                __builtin_prefetch(labelHyp + (size_t)(ct + 2) * BN * DD + tid * 8, 0, 1);
            asm volatile("s_wait_asynccnt 0xc" ::: "memory");
        } else {
            asm volatile("s_wait_asynccnt 0x0" ::: "memory");
        }
        __syncthreads();

        const __bf16* Bs = (ct & 1) ? BsBuf1 : BsBuf0;
        const float yt = yTime[ct * BN + wc * 16 + l16];

        v8f acc = {};
        const __bf16* bRow = Bs + (wc * 16 + l16) * BROW_PAD + half * 8;
#pragma unroll
        for (int kc = 0; kc < NKC; ++kc) {
            v8bf b0 = *(const v8bf*)(bRow + kc * KC);
            v8bf b1 = *(const v8bf*)(bRow + kc * KC + 16);
            v16bf b = __builtin_shufflevector(b0, b1,
                          0,1,2,3,4,5,6,7,8,9,10,11,12,13,14,15);
            acc = __builtin_amdgcn_wmma_f32_16x16x32_bf16(
                false, aFrag[kc], false, b, (short)0, acc, false, false);
        }

        // Epilogue: Lorentz inner = space inner - x_t*y_t; dist is monotone
        // decreasing in it, so hard-negative mining = top-2 largest inners.
        const int col = ct * BN + wc * 16 + l16;
#pragma unroll
        for (int v = 0; v < 8; ++v) {
            float val = acc[v] - xt[v] * yt;
            if (col == tg[v]) {
                posArr[wr * 16 + v + half * 8] = val;   // exactly one writer per row
            } else if (val > max1[v]) {
                max2[v] = max1[v]; max1[v] = val;
            } else if (val > max2[v]) {
                max2[v] = val;
            }
        }
        __syncthreads();   // all waves done with this buffer before it is re-staged
    }

    // Publish per-lane top-2 candidates (32 contributing lanes per row).
#pragma unroll
    for (int v = 0; v < 8; ++v) {
        int rowLocal = wr * 16 + v + half * 8;
        int slot = wc * 16 + l16;
        cand[rowLocal * 64 + slot * 2 + 0] = max1[v];
        cand[rowLocal * 64 + slot * 2 + 1] = max2[v];
    }
    __syncthreads();

    if (tid < BM) {
        float m1 = -3.0e38f, m2 = -3.0e38f;
        const float* c = cand + tid * 64;
        for (int i = 0; i < 64; ++i) {
            float v = c[i];
            if (v > m1) { m2 = m1; m1 = v; }
            else if (v > m2) { m2 = v; }
        }
        const float rsc = rsqrtf(curv);
        // dist(inner) = acosh(max(-curv*inner, 1+eps)) / sqrt(curv)
        float dp = acoshf(fmaxf(-curv * posArr[tid], 1.0f + EPSV)) * rsc;
        float d1 = acoshf(fmaxf(-curv * m1,          1.0f + EPSV)) * rsc;
        float d2 = acoshf(fmaxf(-curv * m2,          1.0f + EPSV)) * rsc;
        float pos = -dp;                                   // CL_TEMP == 1
        float denom = expf(pos) + expf(-d1) + expf(-d2);
        out[rowBase + tid] = logf(denom) - pos;
    }
}

// ---------------- Launch ----------------
extern "C" void kernel_launch(void* const* d_in, const int* in_sizes, int n_in,
                              void* d_out, int out_size, void* d_ws, size_t ws_size,
                              hipStream_t stream) {
    const float* text     = (const float*)d_in[0];
    const float* label    = (const float*)d_in[1];
    const int*   targets  = (const int*)d_in[2];
    const float* curv_log = (const float*)d_in[3];
    const float* ta_log   = (const float*)d_in[4];
    const float* la_log   = (const float*)d_in[5];
    float* out = (float*)d_out;

    char* ws = (char*)d_ws;
    __bf16* textHyp  = (__bf16*)ws;                                  // M*D bf16
    __bf16* labelHyp = (__bf16*)(ws + (size_t)MM * DD * 2);          // C*D bf16
    float*  xTime    = (float*)(ws + (size_t)(MM + CC) * DD * 2);    // M f32
    float*  yTime    = xTime + MM;                                   // C f32

    prep_kernel<<<MM + CC, 256, 0, stream>>>(text, label, curv_log, ta_log, la_log,
                                             textHyp, labelHyp, xTime, yTime);
    cl_loss_kernel<<<MM / BM, 256, 0, stream>>>(textHyp, labelHyp, xTime, yTime,
                                                targets, curv_log, out);
}
